// GATLayer_40381282517195
// MI455X (gfx1250) — matrix operand order repaired and checked by reading, not compiled, not run on hardware
//
#include <hip/hip_runtime.h>

typedef __attribute__((ext_vector_type(16))) _Float16 v16h;
typedef __attribute__((ext_vector_type(8)))  _Float16 v8h;
typedef __attribute__((ext_vector_type(2)))  _Float16 h2;
typedef __attribute__((ext_vector_type(8)))  float    v8f;

extern "C" __device__ _Float16 __ocml_exp2_f16(_Float16);  // -> v_exp_f16

#define NNODES 8192
#define DIM    128
#define HEADS  2
#define HDIM   64
#define SLOPE  0.2f
#define LOG2E  1.4426950408889634f

// -------------------------------------------------------------------------
// Kernel 1: hT = (x @ W)^T in f16.   x:(N,128) f32, W:(128,128) f32.
// Each wave computes a 16x128 row tile with v_wmma_f32_16x16x32_f16.
// Output stored transposed: hT[d][n].
// -------------------------------------------------------------------------
__global__ __launch_bounds__(256) void gat_h_gemm(const float* __restrict__ x,
                                                  const float* __restrict__ W,
                                                  _Float16* __restrict__ hT) {
    __shared__ _Float16 sW[DIM * DIM];   // 32 KB, W as f16, row-major
    const int t = threadIdx.x;
    for (int i = t; i < DIM * DIM; i += 256) sW[i] = (_Float16)W[i];
    __syncthreads();

    const int wave = t >> 5, lane = t & 31;
    const int lm = lane & 15, lh = lane >> 4;
    const int row0 = blockIdx.x * 128 + wave * 16;

    v8f acc[8] = {};

    for (int kb = 0; kb < 4; ++kb) {
        // A tile (16x32 f16): lane = M + 16*(K>>4); half = K&15
        v16h A;
        const float* xp = x + (size_t)(row0 + lm) * DIM + kb * 32 + lh * 16;
        #pragma unroll
        for (int q = 0; q < 8; ++q) {
            A[q]     = (_Float16)xp[q];
            A[8 + q] = (_Float16)xp[8 + q];
        }
        #pragma unroll
        for (int nb = 0; nb < 8; ++nb) {
            // B tile (32x16 f16): lane = N + 16*(K>>4); half = K&15
            v16h B;
            const int krow = kb * 32 + lh * 16;
            const int ncol = nb * 16 + lm;
            #pragma unroll
            for (int v = 0; v < 16; ++v) B[v] = sW[(krow + v) * DIM + ncol];
            acc[nb] = __builtin_amdgcn_wmma_f32_16x16x32_f16(
                false, A, false, B, (short)0, acc[nb], false, false);
        }
    }

    // C layout: VGPR r -> M = r + 8*lh, N = lm. Store transposed.
    #pragma unroll
    for (int nb = 0; nb < 8; ++nb) {
        const int d = nb * 16 + lm;
        _Float16* dst = hT + (size_t)d * NNODES + row0 + 8 * lh;
        v8h pk;
        #pragma unroll
        for (int r = 0; r < 8; ++r) pk[r] = (_Float16)acc[nb][r];
        *(v8h*)dst = pk;
    }
}

// -------------------------------------------------------------------------
// Kernel 2: per-node score halves, packed across heads:
// esp[n] = (e_src_h0[n], e_src_h1[n]) in f16x2; edp likewise for e_dst.
// -------------------------------------------------------------------------
__global__ __launch_bounds__(256) void gat_scores(const _Float16* __restrict__ hT,
                                                  const float* __restrict__ a_src,
                                                  const float* __restrict__ a_dst,
                                                  h2* __restrict__ esp,
                                                  h2* __restrict__ edp) {
    const int n = blockIdx.x * 256 + threadIdx.x;
    float s[HEADS], d[HEADS];
    #pragma unroll
    for (int h = 0; h < HEADS; ++h) {
        float ss = 0.f, dd = 0.f;
        for (int k = 0; k < HDIM; ++k) {
            const float hv = (float)hT[(size_t)(h * HDIM + k) * NNODES + n];
            ss += hv * a_src[h * HDIM + k];
            dd += hv * a_dst[h * HDIM + k];
        }
        s[h] = ss; d[h] = dd;
    }
    h2 e; e.x = (_Float16)s[0]; e.y = (_Float16)s[1];  // lo = head0, hi = head1
    esp[n] = e;
    h2 f; f.x = (_Float16)d[0]; f.y = (_Float16)d[1];
    edp[n] = f;
}

// Packed both-heads score: w = adj>0 ? exp(leaky_relu(es+ed)) : 0, in f16x2.
__device__ __forceinline__ unsigned score_pair(float a, h2 es2, h2 ed2,
                                               h2 slope2, h2 l2e2) {
    h2 t  = es2 + ed2;                                 // v_pk_add_f16
    h2 lr = __builtin_elementwise_max(t, t * slope2);  // leaky relu = max(t,0.2t)
    h2 sc = lr * l2e2;
    h2 w;
    w.x = __ocml_exp2_f16(sc.x);                       // v_exp_f16
    w.y = __ocml_exp2_f16(sc.y);
    union { h2 h; unsigned u; } cv; cv.h = w;
    return a > 0.f ? cv.u : 0u;
}

// -------------------------------------------------------------------------
// Kernel 3: streaming masked-softmax attention + WMMA contraction.
// Block = 256 thr = 8 waves, owns 32 rows (2 row-subtiles of 16).
// Per 32-col step: adj tile (32x32) -> packed f16 w tiles for both heads in
// WMMA-A LDS layout (double buffered, 1 barrier/step, software-pipelined
// global loads). Wave w: head=w>>2, dtile=w&3 -> 2 WMMAs (one per subtile).
// Waves 0/4 also run WMMA vs an all-ones B to get softmax row sums.
// -------------------------------------------------------------------------
__global__ __launch_bounds__(256) void gat_attn(const float* __restrict__ adj,
                                                const _Float16* __restrict__ hT,
                                                const h2* __restrict__ esp,
                                                const h2* __restrict__ edp,
                                                float* __restrict__ out) {
    __shared__ _Float16 sA[2][2][HEADS][16 * 32];  // [buf][subtile][head], 8 KB
    __shared__ float sL[32][HEADS];                // softmax row sums

    const int t = threadIdx.x;
    const int wv = t >> 5, lane = t & 31;
    const int lm = lane & 15, lh = lane >> 4;
    const int ib = blockIdx.x * 32;

    // A-construction role: rows (ai, ai+16), col pair (aj, aj+1)
    const int ai = t >> 4;
    const int aj = (t & 15) * 2;
    const int sl = ai + 16 * (aj >> 4);               // A-layout lane, contiguous k = 16*h + s
    const int sh = aj & 15;                           // contiguous-k A-layout element (even)

    const h2 es2a = esp[ib + ai];
    const h2 es2b = esp[ib + ai + 16];
    h2 slope2; slope2.x = (_Float16)SLOPE; slope2.y = (_Float16)SLOPE;
    h2 l2e2;   l2e2.x   = (_Float16)LOG2E; l2e2.y   = (_Float16)LOG2E;

    // WMMA roles
    const int head  = wv >> 2;
    const int dbase = head * HDIM + (wv & 3) * 16;
    const bool sumwave = (__builtin_amdgcn_readfirstlane(wv) & 3) == 0;
    const _Float16* __restrict__ hrow =
        hT + (size_t)(dbase + lm) * NNODES + lh * 16;  // B-layout base

    const float* adjp0 = adj + (size_t)(ib + ai) * NNODES + aj;
    const float* adjp1 = adjp0 + (size_t)16 * NNODES;

    v16h Bones;
    #pragma unroll
    for (int q = 0; q < 16; ++q) Bones[q] = (_Float16)1.0f;

    v8f acc0 = {}, acc1 = {}, accS0 = {}, accS1 = {};
    int buf = 0;

    // ---- software-pipeline prologue ----
    float2 av0 = *(const float2*)(adjp0);
    float2 av1 = *(const float2*)(adjp1);
    h2     ec0 = edp[aj];
    h2     ec1 = edp[aj + 1];
    v16h   Bc  = *(const v16h*)(hrow);

    for (int jb = 0; jb < NNODES; jb += 32) {
        // issue next step's global loads early (overlap barrier + WMMA)
        const int jn = (jb + 32 < NNODES) ? jb + 32 : 0;
        const float2 av0n = *(const float2*)(adjp0 + jn);
        const float2 av1n = *(const float2*)(adjp1 + jn);
        const h2     ec0n = edp[jn + aj];
        const h2     ec1n = edp[jn + aj + 1];
        const v16h   Bn   = *(const v16h*)(hrow + jn);
        if (jb < NNODES - 1088) {
            __builtin_prefetch(adjp0 + jb + 1024, 0, 1);  // global_prefetch_b8
            __builtin_prefetch(adjp1 + jb + 1024, 0, 1);
        }

        // packed both-heads scores for 4 adj elements (2 rows x 2 cols)
        const unsigned u00 = score_pair(av0.x, es2a, ec0, slope2, l2e2);
        const unsigned u01 = score_pair(av0.y, es2a, ec1, slope2, l2e2);
        const unsigned u10 = score_pair(av1.x, es2b, ec0, slope2, l2e2);
        const unsigned u11 = score_pair(av1.y, es2b, ec1, slope2, l2e2);
        // de-interleave heads: lo16s -> head0 word, hi16s -> head1 word
        *(unsigned*)&sA[buf][0][0][sl * 16 + sh] =
            __builtin_amdgcn_perm(u01, u00, 0x05040100u);
        *(unsigned*)&sA[buf][0][1][sl * 16 + sh] =
            __builtin_amdgcn_perm(u01, u00, 0x07060302u);
        *(unsigned*)&sA[buf][1][0][sl * 16 + sh] =
            __builtin_amdgcn_perm(u11, u10, 0x05040100u);
        *(unsigned*)&sA[buf][1][1][sl * 16 + sh] =
            __builtin_amdgcn_perm(u11, u10, 0x07060302u);

        __syncthreads();  // single barrier per step (double-buffered sA)

        const v16h A0 = *(const v16h*)&sA[buf][0][head][lane * 16];
        const v16h A1 = *(const v16h*)&sA[buf][1][head][lane * 16];

        acc0 = __builtin_amdgcn_wmma_f32_16x16x32_f16(
            false, A0, false, Bc, (short)0, acc0, false, false);
        acc1 = __builtin_amdgcn_wmma_f32_16x16x32_f16(
            false, A1, false, Bc, (short)0, acc1, false, false);
        if (sumwave) {  // scalar branch (readfirstlane) -> EXEC stays all-ones
            accS0 = __builtin_amdgcn_wmma_f32_16x16x32_f16(
                false, A0, false, Bones, (short)0, accS0, false, false);
            accS1 = __builtin_amdgcn_wmma_f32_16x16x32_f16(
                false, A1, false, Bones, (short)0, accS1, false, false);
        }

        av0 = av0n; av1 = av1n; ec0 = ec0n; ec1 = ec1n; Bc = Bn;
        buf ^= 1;
    }

    // Row sums from the ones-WMMA (all 16 lanes of a column identical).
    if (sumwave && lm == 0) {
        #pragma unroll
        for (int r = 0; r < 8; ++r) {
            sL[r + 8 * lh][head]      = accS0[r];
            sL[16 + r + 8 * lh][head] = accS1[r];
        }
    }
    __syncthreads();

    // C layout: VGPR r -> M = r + 8*lh (+16 for subtile 1), N = lm.
    #pragma unroll
    for (int r = 0; r < 8; ++r) {
        const int M0 = r + 8 * lh;
        const int M1 = M0 + 16;
        out[(size_t)(ib + M0) * DIM + dbase + lm] = acc0[r] / sL[M0][head];
        out[(size_t)(ib + M1) * DIM + dbase + lm] = acc1[r] / sL[M1][head];
    }
}

extern "C" void kernel_launch(void* const* d_in, const int* in_sizes, int n_in,
                              void* d_out, int out_size, void* d_ws, size_t ws_size,
                              hipStream_t stream) {
    const float* x     = (const float*)d_in[0];
    const float* adj   = (const float*)d_in[1];
    const float* W     = (const float*)d_in[2];
    const float* a_src = (const float*)d_in[3];
    const float* a_dst = (const float*)d_in[4];
    float* out = (float*)d_out;

    _Float16* hT  = (_Float16*)d_ws;                                     // 2 MB
    h2*       esp = (h2*)((char*)d_ws + (size_t)DIM * NNODES * sizeof(_Float16));
    h2*       edp = esp + NNODES;                                        // 32 KB each

    hipLaunchKernelGGL(gat_h_gemm, dim3(NNODES / 128), dim3(256), 0, stream, x, W, hT);
    hipLaunchKernelGGL(gat_scores, dim3(NNODES / 256), dim3(256), 0, stream,
                       hT, a_src, a_dst, esp, edp);
    hipLaunchKernelGGL(gat_attn, dim3(NNODES / 32), dim3(256), 0, stream,
                       adj, hT, esp, edp, out);
}
